// MultiHeadAttentionLayer_46385646797423
// MI455X (gfx1250) — matrix operand order repaired and checked by reading, not compiled, not run on hardware
//
#include <hip/hip_runtime.h>
#include <hip/hip_bf16.h>

typedef __attribute__((ext_vector_type(16))) _Float16 v16h;
typedef __attribute__((ext_vector_type(8)))  _Float16 v8h;
typedef __attribute__((ext_vector_type(4)))  _Float16 v4h;
typedef __attribute__((ext_vector_type(8)))  float    v8f;

#define B_  4
#define T_  2048
#define H_  1024
#define NH_ 16
#define DH_ 64

// ---------------- WMMA fragment helpers (wave32, gfx1250 layouts) ----------------
// A-matrix 16x32 f16: lane m = l&15; elem e -> k = 16*hb + e
__device__ __forceinline__ v16h frag_a(const _Float16* __restrict__ p, int hb) {
    v8h c0 = *(const v8h*)(p + 16 * hb);
    v8h c1 = *(const v8h*)(p + 16 * hb + 8);
    v16h r;
#pragma unroll
    for (int i = 0; i < 8; i++) { r[i] = c0[i]; r[i + 8] = c1[i]; }
    return r;
}
// B-matrix 32x16 f16 (loaded from transposed N x K storage): lane n = l&15; elem e -> k = 16*hb + e
__device__ __forceinline__ v16h frag_b(const _Float16* __restrict__ p, int hb) {
    v8h c0 = *(const v8h*)(p + 16 * hb);
    v8h c1 = *(const v8h*)(p + 16 * hb + 8);
    v16h r;
#pragma unroll
    for (int i = 0; i < 8; i++) { r[i] = c0[i]; r[i + 8] = c1[i]; }
    return r;
}
__device__ __forceinline__ v8f wmma_f16(v16h a, v16h b, v8f c) {
    return __builtin_amdgcn_wmma_f32_16x16x32_f16(false, a, false, b, (short)0, c,
                                                  false, false);
}

// ---------------- LayerNorm (fp32 in -> f16 out), one block per token ----------------
__global__ void __launch_bounds__(256) ln_kernel(const float* __restrict__ x,
                                                 const float* __restrict__ w,
                                                 const float* __restrict__ bia,
                                                 _Float16* __restrict__ xn) {
    const int t   = blockIdx.x;
    const int tid = threadIdx.x;  // 256 threads * 4 elems = 1024
    const float* row = x + (size_t)t * H_;
    float4 v = ((const float4*)row)[tid];

    float s = v.x + v.y + v.z + v.w;
#pragma unroll
    for (int o = 16; o > 0; o >>= 1) s += __shfl_xor(s, o, 32);
    __shared__ float smem[8];
    int wid = tid >> 5, lid = tid & 31;
    if (lid == 0) smem[wid] = s;
    __syncthreads();
    float tot = 0.f;
#pragma unroll
    for (int i = 0; i < 8; i++) tot += smem[i];
    float mu = tot * (1.0f / H_);

    float dx = v.x - mu, dy = v.y - mu, dz = v.z - mu, dw = v.w - mu;
    float sq = dx * dx + dy * dy + dz * dz + dw * dw;
#pragma unroll
    for (int o = 16; o > 0; o >>= 1) sq += __shfl_xor(sq, o, 32);
    __syncthreads();
    if (lid == 0) smem[wid] = sq;
    __syncthreads();
    float vtot = 0.f;
#pragma unroll
    for (int i = 0; i < 8; i++) vtot += smem[i];
    float inv = rsqrtf(vtot * (1.0f / H_) + 1e-8f);

    float4 wv = ((const float4*)w)[tid];
    float4 bv = ((const float4*)bia)[tid];
    v4h o4;
    o4[0] = (_Float16)(dx * inv * wv.x + bv.x);
    o4[1] = (_Float16)(dy * inv * wv.y + bv.y);
    o4[2] = (_Float16)(dz * inv * wv.z + bv.z);
    o4[3] = (_Float16)(dw * inv * wv.w + bv.w);
    *(v4h*)(xn + (size_t)t * H_ + tid * 4) = o4;
}

// ---------------- Weight transpose + f32->f16: src[K][N] -> dst[N][K] ----------------
__global__ void __launch_bounds__(256) transpose_to_f16(const float* __restrict__ src,
                                                        _Float16* __restrict__ dst,
                                                        int K, int N) {
    size_t i = (size_t)blockIdx.x * blockDim.x + threadIdx.x;
    if (i >= (size_t)K * N) return;
    int k = (int)(i / N), c = (int)(i % N);
    dst[(size_t)c * K + k] = (_Float16)src[i];
}

// ---------------- QKV GEMM: xn[8192,1024] x Wt[3072,1024]^T -> Q,K,Vt ----------------
__global__ void __launch_bounds__(256) gemm_qkv(const _Float16* __restrict__ xn,
                                                const _Float16* __restrict__ wT,
                                                _Float16* __restrict__ Qb,
                                                _Float16* __restrict__ Kb,
                                                _Float16* __restrict__ Vt) {
    int wave = blockIdx.x * (blockDim.x >> 5) + (threadIdx.x >> 5);
    int lane = threadIdx.x & 31;
    int n = lane & 15, hb = lane >> 4;
    const int NG = (3 * H_) / 64;  // 48 column groups of 64
    int mt = wave / NG, ng = wave % NG;
    int row0 = mt * 16, col0 = ng * 64;

    v8f acc[4] = {};
    const _Float16* arow = xn + (size_t)(row0 + n) * H_;
    for (int kb = 0; kb < H_; kb += 32) {
        v16h a = frag_a(arow + kb, hb);
#pragma unroll
        for (int c = 0; c < 4; c++) {
            v16h b = frag_b(wT + (size_t)(col0 + c * 16 + n) * H_ + kb, hb);
            acc[c] = wmma_f16(a, b, acc[c]);
        }
    }

    int sel  = col0 >> 10;            // 0:Q 1:K 2:V (64-col group never straddles)
    int h    = (col0 >> 6) & (NH_ - 1);
    int bidx = row0 / T_;
    int trow = row0 % T_;
    size_t head = (size_t)(bidx * NH_ + h);
#pragma unroll
    for (int c = 0; c < 4; c++) {
        int d = c * 16 + n;
        if (sel == 2) {  // V stored transposed per head: Vt[head][d][t]
            v8h pk;
#pragma unroll
            for (int r = 0; r < 8; r++) pk[r] = (_Float16)acc[c][r];
            *(v8h*)(Vt + (head * DH_ + d) * T_ + trow + 8 * hb) = pk;
        } else {
            _Float16* dstp = (sel == 0 ? Qb : Kb) + (head * T_ + trow) * DH_ + d;
#pragma unroll
            for (int r = 0; r < 8; r++)
                dstp[(size_t)(r + 8 * hb) * DH_] = (_Float16)acc[c][r];
        }
    }
}

// ---------------- Flash attention: one wave = 16 queries of one (b,h) ----------------
__global__ void __launch_bounds__(256) attn_kernel(const _Float16* __restrict__ Qb,
                                                   const _Float16* __restrict__ Kb,
                                                   const _Float16* __restrict__ Vt,
                                                   _Float16* __restrict__ ctx) {
    int wave = blockIdx.x * (blockDim.x >> 5) + (threadIdx.x >> 5);
    int lane = threadIdx.x & 31;
    int n = lane & 15, hb = lane >> 4;
    int qt = wave & (T_ / 16 - 1);
    int h  = (wave >> 7) & (NH_ - 1);
    int b  = wave >> 11;
    size_t head = (size_t)(b * NH_ + h);
    const _Float16* Qh = Qb + head * T_ * DH_;
    const _Float16* Kh = Kb + head * T_ * DH_;
    const _Float16* Vh = Vt + head * DH_ * T_;
    int q0 = qt * 16;

    // Q^T B-fragments for the two d-slices (reused across all key blocks)
    v16h qf0 = frag_b(Qh + (size_t)(q0 + n) * DH_ + 0, hb);
    v16h qf1 = frag_b(Qh + (size_t)(q0 + n) * DH_ + 32, hb);

    v8f acc[4] = {};                 // ctx^T : 4 tiles of (16 d) x (16 q)
    float run_max = -3.0e38f, run_sum = 0.f;
    const float scale = 0.125f;      // 1/sqrt(64)

    for (int kb = 0; kb < T_; kb += 32) {
        const _Float16* k0p = Kh + (size_t)(kb + n) * DH_;
        const _Float16* k1p = Kh + (size_t)(kb + 16 + n) * DH_;
        if (kb + 32 < T_) __builtin_prefetch(k0p + 32 * DH_, 0, 0);

        // S^T = K x Q^T  (two 16-key tiles, K-dim = 64 via two WMMA each)
        v8f s0 = {}, s1 = {};
        s0 = wmma_f16(frag_a(k0p + 0, hb),  qf0, s0);
        s0 = wmma_f16(frag_a(k0p + 32, hb), qf1, s0);
        s1 = wmma_f16(frag_a(k1p + 0, hb),  qf0, s1);
        s1 = wmma_f16(frag_a(k1p + 32, hb), qf1, s1);

        // online softmax along keys (M dim): per-lane reduce + one xor-16 shuffle
        float p0[8], p1[8];
        float bmax = -3.0e38f;
#pragma unroll
        for (int r = 0; r < 8; r++) {
            p0[r] = s0[r] * scale;
            p1[r] = s1[r] * scale;
            bmax = fmaxf(bmax, fmaxf(p0[r], p1[r]));
        }
        bmax = fmaxf(bmax, __shfl_xor(bmax, 16, 32));
        float newmax = fmaxf(run_max, bmax);
        float alpha = __expf(run_max - newmax);
        float lsum = 0.f;
#pragma unroll
        for (int r = 0; r < 8; r++) {
            p0[r] = __expf(p0[r] - newmax);
            p1[r] = __expf(p1[r] - newmax);
            lsum += p0[r] + p1[r];
        }
        lsum += __shfl_xor(lsum, 16, 32);
        run_sum = run_sum * alpha + lsum;
        run_max = newmax;
#pragma unroll
        for (int c = 0; c < 4; c++)
#pragma unroll
            for (int r = 0; r < 8; r++) acc[c][r] *= alpha;

        // reshape probability D-tiles -> B-fragment P (32 keys x 16 queries)
        v16h bp;
#pragma unroll
        for (int e = 0; e < 16; e++) {
            int src = n + ((e >> 3) << 4);
            float v0 = __shfl(p0[e & 7], src, 32);
            float v1 = __shfl(p1[e & 7], src, 32);
            bp[e] = (_Float16)(hb ? v1 : v0);
        }

        // ctx^T += V^T x P
#pragma unroll
        for (int c = 0; c < 4; c++) {
            const _Float16* vp = Vh + (size_t)(c * 16 + n) * T_ + kb;
            acc[c] = wmma_f16(frag_a(vp, hb), bp, acc[c]);
        }
    }

    float inv = 1.f / run_sum;
    int t = q0 + n;
    size_t rowbase = ((size_t)(b * T_ + t)) * H_ + h * DH_;
#pragma unroll
    for (int c = 0; c < 4; c++)
#pragma unroll
        for (int r = 0; r < 8; r++)
            ctx[rowbase + c * 16 + r + 8 * hb] = (_Float16)(acc[c][r] * inv);
}

// ---------------- Output GEMM: ctx[8192,1024] x oT[1024,1024]^T -> f32 out ----------------
__global__ void __launch_bounds__(256) gemm_out(const _Float16* __restrict__ ctx,
                                                const _Float16* __restrict__ oT,
                                                float* __restrict__ out) {
    int wave = blockIdx.x * (blockDim.x >> 5) + (threadIdx.x >> 5);
    int lane = threadIdx.x & 31;
    int n = lane & 15, hb = lane >> 4;
    const int NG = H_ / 64;  // 16
    int mt = wave / NG, ng = wave % NG;
    int row0 = mt * 16, col0 = ng * 64;

    v8f acc[4] = {};
    const _Float16* arow = ctx + (size_t)(row0 + n) * H_;
    for (int kb = 0; kb < H_; kb += 32) {
        v16h a = frag_a(arow + kb, hb);
#pragma unroll
        for (int c = 0; c < 4; c++) {
            v16h b = frag_b(oT + (size_t)(col0 + c * 16 + n) * H_ + kb, hb);
            acc[c] = wmma_f16(a, b, acc[c]);
        }
    }
#pragma unroll
    for (int c = 0; c < 4; c++) {
        float* dp = out + (size_t)(row0 + 8 * hb) * H_ + col0 + c * 16 + n;
#pragma unroll
        for (int r = 0; r < 8; r++) dp[(size_t)r * H_] = acc[c][r];
    }
}

// ---------------- host launcher ----------------
extern "C" void kernel_launch(void* const* d_in, const int* in_sizes, int n_in,
                              void* d_out, int out_size, void* d_ws, size_t ws_size,
                              hipStream_t stream) {
    (void)in_sizes; (void)n_in; (void)out_size; (void)ws_size;
    const float* x   = (const float*)d_in[0];
    const float* lnw = (const float*)d_in[1];
    const float* lnb = (const float*)d_in[2];
    const float* qkv = (const float*)d_in[3];
    const float* o   = (const float*)d_in[4];
    float* out = (float*)d_out;

    const size_t NTOK = (size_t)B_ * T_;  // 8192
    _Float16* ws   = (_Float16*)d_ws;
    _Float16* xn   = ws;  ws += NTOK * H_;
    _Float16* qkvT = ws;  ws += (size_t)3 * H_ * H_;
    _Float16* oT   = ws;  ws += (size_t)H_ * H_;
    _Float16* Qb   = ws;  ws += NTOK * H_;
    _Float16* Kb   = ws;  ws += NTOK * H_;
    _Float16* Vt   = ws;  ws += NTOK * H_;
    _Float16* ctx  = ws;  ws += NTOK * H_;

    ln_kernel<<<(unsigned)NTOK, 256, 0, stream>>>(x, lnw, lnb, xn);
    transpose_to_f16<<<(3 * H_ * H_) / 256, 256, 0, stream>>>(qkv, qkvT, H_, 3 * H_);
    transpose_to_f16<<<(H_ * H_) / 256, 256, 0, stream>>>(o, oT, H_, H_);
    gemm_qkv<<<((NTOK / 16) * (3 * H_ / 64)) / 8, 256, 0, stream>>>(xn, qkvT, Qb, Kb, Vt);
    attn_kernel<<<(B_ * NH_ * (T_ / 16)) / 8, 256, 0, stream>>>(Qb, Kb, Vt, ctx);
    gemm_out<<<((NTOK / 16) * (H_ / 64)) / 8, 256, 0, stream>>>(ctx, oT, out);
}
